// LGEB_8366596292936
// MI455X (gfx1250) — compile-verified, matrix operand order checked by reading, not run on hardware
//
#include <hip/hip_runtime.h>
#include <hip/hip_bf16.h>

// LGEB (LorentzNet block) for MI455X / gfx1250, wave32 + WMMA f32<-f16.
//
// Pipeline (all on `stream`):
//   zero_ws                    : stats/magg/agg/cnt accumulators
//   swizzle_w x5               : weights -> f16 fragment-major layout in ws
//   edge_pass1  (4096 blocks)  : z1 = feat@We1 (WMMA), BN stats (sum, sum^2)
//   finalize_bn                : scale/shift for edge BN
//   edge_pass2  (4096 blocks)  : recompute z1, BN+ReLU, @We2+ReLU, sigmoid
//                                gate, write m, atomics magg/agg/cnt, phi_x
//   node_pass1  (32 blocks)    : z2 = [h,magg,attr]@Wh1+bh1 (WMMA), BN stats
//   finalize_bn                : scale/shift for node BN
//   node_pass2  (32 blocks)    : BN+ReLU, @Wh2+bh2, h_out = h + ..., x_out

typedef __attribute__((ext_vector_type(16))) _Float16 v16h;
typedef __attribute__((ext_vector_type(8)))  _Float16 v8h;
typedef __attribute__((ext_vector_type(8)))  float    v8f;

#define B_   32
#define N_   128
#define E_   16384
#define D_   72
#define H_   72
#define A_   4
#define EPS_ 1e-5f

#define KE   160   // padded K for 2D+2 = 146
#define KH   96    // padded K for 72
#define NP   80    // padded N for 72

// ---- workspace layout (bytes) ----
#define OFF_WE1  ((size_t)0)                 // 5x5 tiles * 512 halves = 25600 B
#define OFF_WE2  (OFF_WE1 + 25600)           // 3x5 tiles = 15360 B
#define OFF_WX1  (OFF_WE2 + 15360)
#define OFF_WH1  (OFF_WX1 + 15360)           // 5x5 = 25600 B
#define OFF_WH2  (OFF_WH1 + 25600)           // 3x5 = 15360 B
#define OFF_F32  (OFF_WH2 + 15360)           // = 97280, 256B aligned

// ---------------- fragment loaders (ISA 7.12.2 16-bit layouts) -------------

__device__ __forceinline__ v16h load_a_frag(const _Float16* lds, int rowBase,
                                            int ld, int kBase) {
  int lane = threadIdx.x & 31;
  const _Float16* p = lds + (size_t)(rowBase + (lane & 15)) * ld + kBase +
                      ((lane >> 4) << 3);
  v8h lo = *(const v8h*)p;          // K = kBase+koff .. +7
  v8h hi = *(const v8h*)(p + 16);   // K = kBase+koff+16 .. +23
  v16h r;
#pragma unroll
  for (int e = 0; e < 8; ++e) { r[e] = lo[e]; r[e + 8] = hi[e]; }
  return r;
}

// B matrix stored fragment-major by swizzle_w: one contiguous 32B per lane.
__device__ __forceinline__ v16h load_b_frag(const _Float16* w, int kt, int nt,
                                            int nTiles) {
  int lane = threadIdx.x & 31;
  return *(const v16h*)(w + (((size_t)(kt * nTiles + nt)) * 32 + lane) * 16);
}

template <int KT, int NT>
__device__ __forceinline__ void gemm_tile(const _Float16* ldsA, int ld,
                                          int rowBase, const _Float16* Wsw,
                                          v8f acc[NT]) {
#pragma unroll
  for (int kt = 0; kt < KT; ++kt) {
    v16h a = load_a_frag(ldsA, rowBase, ld, kt * 32);
#pragma unroll
    for (int nt = 0; nt < NT; ++nt) {
      v16h b = load_b_frag(Wsw, kt, nt, NT);
      acc[nt] = __builtin_amdgcn_wmma_f32_16x16x32_f16(
          false, a, false, b, (short)0, acc[nt], false, false);
    }
  }
}

__device__ __forceinline__ float psi_(float v) {
  return copysignf(log1pf(fabsf(v)), v);
}

// --------------------------- prep kernels ----------------------------------

__global__ void zero_ws_k(float* p, size_t n) {
  for (size_t i = (size_t)blockIdx.x * blockDim.x + threadIdx.x; i < n;
       i += (size_t)gridDim.x * blockDim.x)
    p[i] = 0.f;
}

// src: row-major [Ks, Ns] f32; dst: fragment-major padded f16.
__global__ void swizzle_w(const float* __restrict__ src, _Float16* __restrict__ dst,
                          int Ks, int Ns, int KT, int NT) {
  int total = KT * NT * 512;
  for (int idx = blockIdx.x * blockDim.x + threadIdx.x; idx < total;
       idx += gridDim.x * blockDim.x) {
    int e    = idx & 15;
    int lane = (idx >> 4) & 31;
    int tile = idx >> 9;
    int nt = tile % NT, kt = tile / NT;
    int k = kt * 32 + ((lane >> 4) << 3) + e + ((e >= 8) ? 8 : 0);
    int n = nt * 16 + (lane & 15);
    float v = (k < Ks && n < Ns) ? src[(size_t)k * Ns + n] : 0.f;
    dst[idx] = (_Float16)v;
  }
}

__global__ void finalize_bn(const float* __restrict__ sum, const float* __restrict__ sq,
                            const float* __restrict__ g, const float* __restrict__ bb,
                            float* __restrict__ scale, float* __restrict__ shift,
                            float invCount) {
  int c = threadIdx.x;
  if (c >= NP) return;
  float mu  = sum[c] * invCount;
  float var = sq[c] * invCount - mu * mu;
  float rs  = rsqrtf(var + EPS_);
  float gg  = (c < H_) ? g[c] : 0.f;   // pad channels -> identically zero
  float bv  = (c < H_) ? bb[c] : 0.f;
  scale[c] = gg * rs;
  shift[c] = bv - mu * gg * rs;
}

// --------------------------- edge feature gather ---------------------------

__device__ __forceinline__ void gather_edge(const float* __restrict__ h,
                                            const float* __restrict__ x,
                                            const int* __restrict__ ei,
                                            const int* __restrict__ ej,
                                            int b, int e0, _Float16* sF,
                                            int* sIdx, float* sXd) {
  int tid = threadIdx.x;
  int r = tid >> 1, half = tid & 1;
  int e = e0 + r;
  int i = ei[(size_t)b * E_ + e];
  int j = ej[(size_t)b * E_ + e];
  if (half == 0) {
    const float* hr = h + ((size_t)b * N_ + i) * D_;
#pragma unroll 8
    for (int c = 0; c < D_; ++c) sF[(size_t)r * KE + c] = (_Float16)hr[c];
    const float* xi = x + ((size_t)b * N_ + i) * 4;
    const float* xj = x + ((size_t)b * N_ + j) * 4;
    float d0 = xi[0] - xj[0], d1 = xi[1] - xj[1];
    float d2 = xi[2] - xj[2], d3 = xi[3] - xj[3];
    float nrm = d0 * d0 - d1 * d1 - d2 * d2 - d3 * d3;       // Minkowski |dx|^2
    float dot = xi[0] * xj[0] - xi[1] * xj[1] - xi[2] * xj[2] - xi[3] * xj[3];
    sF[(size_t)r * KE + 144] = (_Float16)psi_(nrm);
    sF[(size_t)r * KE + 145] = (_Float16)psi_(dot);
#pragma unroll
    for (int c = 146; c < KE; ++c) sF[(size_t)r * KE + c] = (_Float16)0.f;
    if (sIdx) {
      sIdx[r] = i;
      sXd[r * 4 + 0] = d0; sXd[r * 4 + 1] = d1;
      sXd[r * 4 + 2] = d2; sXd[r * 4 + 3] = d3;
    }
  } else {
    const float* hr = h + ((size_t)b * N_ + j) * D_;
#pragma unroll 8
    for (int c = 0; c < D_; ++c) sF[(size_t)r * KE + 72 + c] = (_Float16)hr[c];
  }
}

// --------------------------- edge pass 1 (BN stats) ------------------------

__global__ __launch_bounds__(256) void edge_pass1(
    const float* __restrict__ h, const float* __restrict__ x,
    const int* __restrict__ ei, const int* __restrict__ ej,
    const _Float16* __restrict__ We1sw, float* sumE, float* sqE) {
  __shared__ _Float16 sF[128 * KE];
  __shared__ float sSum[NP], sSq[NP];
  int blk = blockIdx.x;
  int b = blk >> 7, e0 = (blk & 127) * 128;
  int tid = threadIdx.x;
  if (tid < NP) { sSum[tid] = 0.f; sSq[tid] = 0.f; }
  gather_edge(h, x, ei, ej, b, e0, sF, nullptr, nullptr);
  __syncthreads();

  int wave = tid >> 5, lane = tid & 31;
  v8f acc[5] = {};
  gemm_tile<5, 5>(sF, KE, wave * 16, We1sw, acc);

#pragma unroll
  for (int nt = 0; nt < 5; ++nt) {
    float s = 0.f, q = 0.f;
#pragma unroll
    for (int v = 0; v < 8; ++v) { float a = acc[nt][v]; s += a; q += a * a; }
    int col = nt * 16 + (lane & 15);
    atomicAdd(&sSum[col], s);
    atomicAdd(&sSq[col], q);
  }
  __syncthreads();
  if (tid < NP) { atomicAdd(&sumE[tid], sSum[tid]); atomicAdd(&sqE[tid], sSq[tid]); }
}

// --------------------------- edge pass 2 (fused chain) ---------------------

__global__ __launch_bounds__(256) void edge_pass2(
    const float* __restrict__ h, const float* __restrict__ x,
    const int* __restrict__ ei, const int* __restrict__ ej,
    const _Float16* __restrict__ We1sw, const _Float16* __restrict__ We2sw,
    const _Float16* __restrict__ Wx1sw,
    const float* __restrict__ scaleE, const float* __restrict__ shiftE,
    const float* __restrict__ be2, const float* __restrict__ Wm,
    const float* __restrict__ bm, const float* __restrict__ bx1,
    const float* __restrict__ Wx2,
    float* __restrict__ m_out, float* magg, float* agg, float* cnt) {
  __shared__ _Float16 sF[128 * KE];
  __shared__ int   sIdx[128];
  __shared__ float sXd[128 * 4];
  int blk = blockIdx.x;
  int b = blk >> 7, e0 = (blk & 127) * 128;
  int tid = threadIdx.x;
  gather_edge(h, x, ei, ej, b, e0, sF, sIdx, sXd);
  __syncthreads();

  int wave = tid >> 5, lane = tid & 31;
  int rowBase = wave * 16;
  int colLo = lane & 15;
  int rowOff = (lane >> 4) << 3;   // 0 or 8, per C-fragment layout

  // z1 = feat @ We1
  v8f acc1[5] = {};
  gemm_tile<5, 5>(sF, KE, rowBase, We1sw, acc1);

  // BN scale/shift + ReLU -> a1, staged back into the (wave-local) LDS rows.
  // Pad cols 72..79 get scale=shift=0 => exact zeros.
#pragma unroll
  for (int nt = 0; nt < 5; ++nt) {
    int col = nt * 16 + colLo;
    float sc = scaleE[col], sh = shiftE[col];
#pragma unroll
    for (int v = 0; v < 8; ++v) {
      float a = fmaxf(acc1[nt][v] * sc + sh, 0.f);
      int row = rowBase + v + rowOff;
      sF[(size_t)row * KE + col] = (_Float16)a;
    }
  }
  { // zero K-pad cols 80..95 of this wave's rows
    int row = rowBase + (lane & 15);
    int c0 = 80 + ((lane >> 4) << 3);
#pragma unroll
    for (int c = 0; c < 8; ++c) sF[(size_t)row * KE + c0 + c] = (_Float16)0.f;
  }

  // z = a1 @ We2 + be2 ; ReLU -> mval ; dot with Wm for sigmoid gate
  v8f acc2[5] = {};
  gemm_tile<3, 5>(sF, KE, rowBase, We2sw, acc2);
  float dotm[8] = {0, 0, 0, 0, 0, 0, 0, 0};
#pragma unroll
  for (int nt = 0; nt < 5; ++nt) {
    int col = nt * 16 + colLo;
    float bias = (col < H_) ? be2[col] : 0.f;
    float wm   = (col < H_) ? Wm[col]  : 0.f;
#pragma unroll
    for (int v = 0; v < 8; ++v) {
      float mv = fmaxf(acc2[nt][v] + bias, 0.f);
      acc2[nt][v] = mv;
      dotm[v] += mv * wm;
    }
  }
  float bm0 = bm[0];
#pragma unroll
  for (int v = 0; v < 8; ++v) {
#pragma unroll
    for (int off = 8; off > 0; off >>= 1) dotm[v] += __shfl_xor(dotm[v], off, 16);
    dotm[v] = 1.f / (1.f + __expf(-(dotm[v] + bm0)));       // sigmoid gate
  }

  // gated message: store to d_out, atomically segment-sum into magg,
  // and restage as f16 for the phi_x GEMM.
  size_t mBase = ((size_t)b * E_ + e0) * (size_t)H_;
#pragma unroll
  for (int nt = 0; nt < 5; ++nt) {
    int col = nt * 16 + colLo;
#pragma unroll
    for (int v = 0; v < 8; ++v) {
      float mg = acc2[nt][v] * dotm[v];
      int row = rowBase + v + rowOff;
      if (col < H_) {
        m_out[mBase + (size_t)row * H_ + col] = mg;
        int node = sIdx[row];
        atomicAdd(&magg[((size_t)b * N_ + node) * H_ + col], mg);
      }
      sF[(size_t)row * KE + col] = (_Float16)mg;
    }
  }

  // phi_x: g = ReLU(m @ Wx1 + bx1); gate = g @ Wx2
  v8f acc3[5] = {};
  gemm_tile<3, 5>(sF, KE, rowBase, Wx1sw, acc3);
  float dotx[8] = {0, 0, 0, 0, 0, 0, 0, 0};
#pragma unroll
  for (int nt = 0; nt < 5; ++nt) {
    int col = nt * 16 + colLo;
    float bias = (col < H_) ? bx1[col] : 0.f;
    float wx   = (col < H_) ? Wx2[col] : 0.f;
#pragma unroll
    for (int v = 0; v < 8; ++v) {
      float g = fmaxf(acc3[nt][v] + bias, 0.f);
      dotx[v] += g * wx;
    }
  }
#pragma unroll
  for (int v = 0; v < 8; ++v) {
#pragma unroll
    for (int off = 8; off > 0; off >>= 1) dotx[v] += __shfl_xor(dotx[v], off, 16);
  }
  if (colLo == 0) {
#pragma unroll
    for (int v = 0; v < 8; ++v) {
      int row = rowBase + v + rowOff;
      int node = sIdx[row];
      float gate = dotx[v];
      size_t ab = ((size_t)b * N_ + node) * 4;
#pragma unroll
      for (int c = 0; c < 4; ++c) {
        float t = fminf(fmaxf(sXd[row * 4 + c] * gate, -100.f), 100.f);
        atomicAdd(&agg[ab + c], t);
      }
      atomicAdd(&cnt[(size_t)b * N_ + node], 1.f);
    }
  }
}

// --------------------------- node pass 1 -----------------------------------

__global__ __launch_bounds__(256) void node_pass1(
    const float* __restrict__ h, const float* __restrict__ magg,
    const float* __restrict__ attr, const _Float16* __restrict__ Wh1sw,
    const float* __restrict__ bh1, float* __restrict__ zst,
    float* sumH, float* sqH) {
  __shared__ _Float16 sF[128 * KE];
  __shared__ float sSum[NP], sSq[NP];
  int tid = threadIdx.x;
  if (tid < NP) { sSum[tid] = 0.f; sSq[tid] = 0.f; }
  int r = tid >> 1, half = tid & 1;
  size_t gr = (size_t)blockIdx.x * 128 + r;
  if (half == 0) {
    const float* hr = h + gr * D_;
#pragma unroll 8
    for (int c = 0; c < D_; ++c) sF[(size_t)r * KE + c] = (_Float16)hr[c];
    const float* ar = attr + gr * 4;
#pragma unroll
    for (int c = 0; c < A_; ++c) sF[(size_t)r * KE + 144 + c] = (_Float16)ar[c];
#pragma unroll
    for (int c = 148; c < KE; ++c) sF[(size_t)r * KE + c] = (_Float16)0.f;
  } else {
    const float* mr = magg + gr * H_;
#pragma unroll 8
    for (int c = 0; c < H_; ++c) sF[(size_t)r * KE + 72 + c] = (_Float16)mr[c];
  }
  __syncthreads();

  int wave = tid >> 5, lane = tid & 31;
  int rowBase = wave * 16, colLo = lane & 15, rowOff = (lane >> 4) << 3;
  v8f acc[5] = {};
  gemm_tile<5, 5>(sF, KE, rowBase, Wh1sw, acc);
  size_t grBase = (size_t)blockIdx.x * 128;
#pragma unroll
  for (int nt = 0; nt < 5; ++nt) {
    int col = nt * 16 + colLo;
    float bias = (col < H_) ? bh1[col] : 0.f;
    float s = 0.f, q = 0.f;
#pragma unroll
    for (int v = 0; v < 8; ++v) {
      float z = acc[nt][v] + bias;
      int row = rowBase + v + rowOff;
      zst[(grBase + row) * NP + col] = z;
      s += z; q += z * z;
    }
    atomicAdd(&sSum[col], s);
    atomicAdd(&sSq[col], q);
  }
  __syncthreads();
  if (tid < NP) { atomicAdd(&sumH[tid], sSum[tid]); atomicAdd(&sqH[tid], sSq[tid]); }
}

// --------------------------- node pass 2 -----------------------------------

__global__ __launch_bounds__(256) void node_pass2(
    const float* __restrict__ zst, const float* __restrict__ scaleH,
    const float* __restrict__ shiftH, const _Float16* __restrict__ Wh2sw,
    const float* __restrict__ bh2, const float* __restrict__ h,
    const float* __restrict__ x, const float* __restrict__ agg,
    const float* __restrict__ cnt, float* __restrict__ h_out,
    float* __restrict__ x_out) {
  __shared__ _Float16 sA[128 * KH];
  int tid = threadIdx.x;
  size_t grBase = (size_t)blockIdx.x * 128;
  for (int idx = tid; idx < 128 * KH; idx += 256) {
    int r = idx / KH, c = idx % KH;
    float v = 0.f;
    if (c < NP) {
      float z = zst[(grBase + r) * NP + c];
      v = fmaxf(z * scaleH[c] + shiftH[c], 0.f);
    }
    sA[idx] = (_Float16)v;
  }
  __syncthreads();

  int wave = tid >> 5, lane = tid & 31;
  int rowBase = wave * 16, colLo = lane & 15, rowOff = (lane >> 4) << 3;
  v8f acc[5] = {};
  gemm_tile<3, 5>(sA, KH, rowBase, Wh2sw, acc);
#pragma unroll
  for (int nt = 0; nt < 5; ++nt) {
    int col = nt * 16 + colLo;
    if (col < H_) {
      float bias = bh2[col];
#pragma unroll
      for (int v = 0; v < 8; ++v) {
        int row = rowBase + v + rowOff;
        size_t gi = (grBase + row) * (size_t)H_ + col;
        h_out[gi] = h[gi] + acc[nt][v] + bias;      // residual
      }
    }
  }
  if (tid < 128) {
    size_t gr = grBase + tid;
    float ct = fmaxf(cnt[gr], 1.f);
#pragma unroll
    for (int c = 0; c < 4; ++c)
      x_out[gr * 4 + c] = x[gr * 4 + c] + agg[gr * 4 + c] / ct;  // C_WEIGHT=1
  }
}

// --------------------------- launcher --------------------------------------

extern "C" void kernel_launch(void* const* d_in, const int* in_sizes, int n_in,
                              void* d_out, int out_size, void* d_ws, size_t ws_size,
                              hipStream_t stream) {
  (void)in_sizes; (void)n_in; (void)out_size; (void)ws_size;
  const float* h    = (const float*)d_in[0];
  const float* x    = (const float*)d_in[1];
  const int*   ei   = (const int*)d_in[2];
  const int*   ej   = (const int*)d_in[3];
  const float* attr = (const float*)d_in[4];
  const float* We1  = (const float*)d_in[5];
  const float* g1   = (const float*)d_in[6];
  const float* b1   = (const float*)d_in[7];
  const float* We2  = (const float*)d_in[8];
  const float* be2  = (const float*)d_in[9];
  const float* Wm   = (const float*)d_in[10];
  const float* bm   = (const float*)d_in[11];
  const float* Wx1  = (const float*)d_in[12];
  const float* bx1  = (const float*)d_in[13];
  const float* Wx2  = (const float*)d_in[14];
  const float* Wh1  = (const float*)d_in[15];
  const float* bh1  = (const float*)d_in[16];
  const float* gh   = (const float*)d_in[17];
  const float* bh   = (const float*)d_in[18];
  const float* Wh2  = (const float*)d_in[19];
  const float* bh2  = (const float*)d_in[20];

  char* ws = (char*)d_ws;
  _Float16* We1sw = (_Float16*)(ws + OFF_WE1);
  _Float16* We2sw = (_Float16*)(ws + OFF_WE2);
  _Float16* Wx1sw = (_Float16*)(ws + OFF_WX1);
  _Float16* Wh1sw = (_Float16*)(ws + OFF_WH1);
  _Float16* Wh2sw = (_Float16*)(ws + OFF_WH2);

  float* f32    = (float*)(ws + OFF_F32);
  float* sumE   = f32;          float* sqE    = f32 + 80;
  float* sumH   = f32 + 160;    float* sqH    = f32 + 240;
  float* scaleE = f32 + 320;    float* shiftE = f32 + 400;
  float* scaleH = f32 + 480;    float* shiftH = f32 + 560;
  float* magg   = f32 + 640;                       // [B*N, 72]
  float* agg    = magg + (size_t)B_ * N_ * H_;     // [B*N, 4]
  float* cnt    = agg  + (size_t)B_ * N_ * 4;      // [B*N]
  float* zst    = cnt  + (size_t)B_ * N_;          // [B*N, NP]

  float* h_out = (float*)d_out;
  float* x_out = h_out + (size_t)B_ * N_ * H_;
  float* m_out = x_out + (size_t)B_ * N_ * 4;

  // zero stats + magg + agg + cnt every launch (harness does not re-zero ws)
  size_t nz = 640 + (size_t)B_ * N_ * (H_ + 4 + 1);
  zero_ws_k<<<256, 256, 0, stream>>>(f32, nz);

  swizzle_w<<<32, 256, 0, stream>>>(We1, We1sw, 2 * D_ + 2, H_, 5, 5);
  swizzle_w<<<32, 256, 0, stream>>>(We2, We2sw, H_, H_, 3, 5);
  swizzle_w<<<32, 256, 0, stream>>>(Wx1, Wx1sw, H_, H_, 3, 5);
  swizzle_w<<<32, 256, 0, stream>>>(Wh1, Wh1sw, H_ + D_ + A_, H_, 5, 5);
  swizzle_w<<<32, 256, 0, stream>>>(Wh2, Wh2sw, H_, H_, 3, 5);

  edge_pass1<<<B_ * (E_ / 128), 256, 0, stream>>>(h, x, ei, ej, We1sw, sumE, sqE);
  finalize_bn<<<1, 128, 0, stream>>>(sumE, sqE, g1, b1, scaleE, shiftE,
                                     1.f / ((float)B_ * (float)E_));
  edge_pass2<<<B_ * (E_ / 128), 256, 0, stream>>>(
      h, x, ei, ej, We1sw, We2sw, Wx1sw, scaleE, shiftE, be2, Wm, bm, bx1, Wx2,
      m_out, magg, agg, cnt);

  node_pass1<<<(B_ * N_) / 128, 256, 0, stream>>>(h, magg, attr, Wh1sw, bh1,
                                                  zst, sumH, sqH);
  finalize_bn<<<1, 128, 0, stream>>>(sumH, sqH, gh, bh, scaleH, shiftH,
                                     1.f / ((float)B_ * (float)N_));
  node_pass2<<<(B_ * N_) / 128, 256, 0, stream>>>(
      zst, scaleH, shiftH, Wh2sw, bh2, h, x, agg, cnt, h_out, x_out);
}
